// ABMIL_90546500534715
// MI455X (gfx1250) — compile-verified, matrix operand order checked by reading, not run on hardware
//
#include <hip/hip_runtime.h>
#include <hip/hip_bf16.h>

// ---------------- problem constants ----------------
static constexpr int B   = 4;
static constexpr int N0  = 2048;
static constexpr int S   = 2049;        // N0 + cls
static constexpr int SP  = 2080;        // S padded to multiple of 32 (key dim)
static constexpr int QT  = (S + 15) / 16;   // 129 query tiles
static constexpr int D   = 384;
static constexpr int TD  = 1152;        // 3*D
static constexpr int NH  = 6;
static constexpr int HD  = 64;
static constexpr int L   = 2;
static constexpr int TOPK = 16;
static constexpr int MS  = B * S;       // 8196 rows (region)
static constexpr int MT  = B * TOPK;    // 64 rows  (token)
static constexpr int GK  = D;           // GEMM K is always 384
static constexpr int KST = GK / 32;     // 12 k-steps

typedef __bf16 v16bf __attribute__((ext_vector_type(16)));
typedef __bf16 v8bf  __attribute__((ext_vector_type(8)));
typedef float  v8f   __attribute__((ext_vector_type(8)));

// ---------------- WMMA helpers ----------------
__device__ __forceinline__ v8f wmma_bf16(v16bf a, v16bf b, v8f c) {
  // (neg_a, A, neg_b, B, c_mod, C, reuse_a, reuse_b)
  return __builtin_amdgcn_wmma_f32_16x16x32_bf16(false, a, false, b, (short)0, c, false, false);
}

// Load a 16x32 A/B fragment row: per-lane 16 bf16 elements,
// k(j) = j + 8*(j>=8) + 8*hi  -> two contiguous 16B loads.
__device__ __forceinline__ v16bf load_frag(const __bf16* rowp, int kc, int hi) {
  v8bf a = *reinterpret_cast<const v8bf*>(rowp + kc + 8 * hi);
  v8bf b = *reinterpret_cast<const v8bf*>(rowp + kc + 16 + 8 * hi);
  v16bf r;
#pragma unroll
  for (int j = 0; j < 8; ++j) { r[j] = a[j]; r[j + 8] = b[j]; }
  return r;
}

// ---------------- elementwise / prep kernels ----------------
__global__ void k_build_y(const float* __restrict__ x, const float* __restrict__ cls,
                          float* __restrict__ y) {
  long long i = (long long)blockIdx.x * 256 + threadIdx.x;
  long long total = (long long)B * S * D;
  if (i >= total) return;
  int d = (int)(i % D);
  long long rs = i / D;
  int s = (int)(rs % S);
  int b = (int)(rs / S);
  y[i] = (s == 0) ? cls[d] : x[((long long)b * N0 + (s - 1)) * D + d];
}

__global__ void k_cvt_bf16(const float* __restrict__ in, __bf16* __restrict__ out, long long n) {
  long long i = (long long)blockIdx.x * 256 + threadIdx.x;
  if (i < n) out[i] = (__bf16)in[i];
}

// LayerNorm over D=384, optional residual (indexed by dest row) and row gather (for `in`).
__global__ void k_ln(const float* __restrict__ in, const float* __restrict__ resid,
                     const int* __restrict__ gather, const float* __restrict__ gam,
                     const float* __restrict__ bet, float* __restrict__ outf,
                     __bf16* __restrict__ outb) {
  int row = blockIdx.x;
  int tid = threadIdx.x;                       // 128 threads
  long long src = gather ? (long long)gather[row] : (long long)row;
  const float* hp = in + src * (long long)D;
  float v[3];
#pragma unroll
  for (int i = 0; i < 3; ++i) {
    int d = tid + i * 128;
    float t = hp[d];
    if (resid) t += resid[(long long)row * D + d];
    v[i] = t;
  }
  __shared__ float red[128];
  red[tid] = v[0] + v[1] + v[2];
  __syncthreads();
  for (int st = 64; st > 0; st >>= 1) { if (tid < st) red[tid] += red[tid + st]; __syncthreads(); }
  float mean = red[0] * (1.f / D);
  __syncthreads();
  float q = 0.f;
#pragma unroll
  for (int i = 0; i < 3; ++i) { float dd = v[i] - mean; q += dd * dd; }
  red[tid] = q;
  __syncthreads();
  for (int st = 64; st > 0; st >>= 1) { if (tid < st) red[tid] += red[tid + st]; __syncthreads(); }
  float rstd = rsqrtf(red[0] * (1.f / D) + 1e-5f);
#pragma unroll
  for (int i = 0; i < 3; ++i) {
    int d = tid + i * 128;
    float o = (v[i] - mean) * rstd * gam[d] + bet[d];
    long long off = (long long)row * D + d;
    if (outf) outf[off] = o;
    if (outb) outb[off] = (__bf16)o;
  }
}

// ---------------- GEMM: out[M,N] = A[M,K=384] @ W[N,K]^T + bias ----------------
// block = 128 thr (4 waves); wave -> 32(M) x 64(N) tile (8 C tiles),
// software-pipelined fragment loads so WMMAs overlap VMEM.
__global__ void k_gemm(const __bf16* __restrict__ A, const __bf16* __restrict__ W,
                       const float* __restrict__ bias, float* __restrict__ outf,
                       __bf16* __restrict__ outb, int M, int N) {
  int lane = threadIdx.x & 31;
  int wv   = threadIdx.x >> 5;
  int lr   = lane & 15;
  int hi   = lane >> 4;
  int m0   = blockIdx.x * 128 + wv * 32;
  int n0   = blockIdx.y * 64;
  int ar0  = m0 + lr;      if (ar0 > M - 1) ar0 = M - 1;
  int ar1  = m0 + 16 + lr; if (ar1 > M - 1) ar1 = M - 1;
  const __bf16* ap0 = A + (long long)ar0 * GK;
  const __bf16* ap1 = A + (long long)ar1 * GK;
  const __bf16* wp0 = W + (long long)(n0 + 0 * 16 + lr) * GK;
  const __bf16* wp1 = W + (long long)(n0 + 1 * 16 + lr) * GK;
  const __bf16* wp2 = W + (long long)(n0 + 2 * 16 + lr) * GK;
  const __bf16* wp3 = W + (long long)(n0 + 3 * 16 + lr) * GK;

  v8f acc[8] = {};
  v16bf a0 = load_frag(ap0, 0, hi);
  v16bf a1 = load_frag(ap1, 0, hi);
  v16bf b0 = load_frag(wp0, 0, hi);
  v16bf b1 = load_frag(wp1, 0, hi);
  v16bf b2 = load_frag(wp2, 0, hi);
  v16bf b3 = load_frag(wp3, 0, hi);

#pragma unroll
  for (int ks = 0; ks < KST; ++ks) {
    v16bf na0, na1, nb0, nb1, nb2, nb3;
    if (ks + 1 < KST) {               // prefetch next k-step's fragments
      int kc = (ks + 1) * 32;
      na0 = load_frag(ap0, kc, hi);
      na1 = load_frag(ap1, kc, hi);
      nb0 = load_frag(wp0, kc, hi);
      nb1 = load_frag(wp1, kc, hi);
      nb2 = load_frag(wp2, kc, hi);
      nb3 = load_frag(wp3, kc, hi);
    }
    acc[0] = wmma_bf16(a0, b0, acc[0]);
    acc[1] = wmma_bf16(a0, b1, acc[1]);
    acc[2] = wmma_bf16(a0, b2, acc[2]);
    acc[3] = wmma_bf16(a0, b3, acc[3]);
    acc[4] = wmma_bf16(a1, b0, acc[4]);
    acc[5] = wmma_bf16(a1, b1, acc[5]);
    acc[6] = wmma_bf16(a1, b2, acc[6]);
    acc[7] = wmma_bf16(a1, b3, acc[7]);
    a0 = na0; a1 = na1; b0 = nb0; b1 = nb1; b2 = nb2; b3 = nb3;
  }

#pragma unroll
  for (int half = 0; half < 2; ++half)
#pragma unroll
    for (int t = 0; t < 4; ++t) {
      int n = n0 + t * 16 + lr;
      float bv = bias ? bias[n] : 0.f;
      v8f c = acc[half * 4 + t];
#pragma unroll
      for (int j = 0; j < 8; ++j) {
        int m = m0 + half * 16 + j + 8 * hi;
        if (m < M) {
          float v = c[j] + bv;
          long long off = (long long)m * N + n;
          if (outf) outf[off] = v;
          if (outb) outb[off] = (__bf16)v;
        }
      }
    }
}

// ---------------- V transpose: Vt[b,h,d,s] (bf16, keys padded to SP w/ 0) ----------------
__global__ void k_vtr(const __bf16* __restrict__ qkv, __bf16* __restrict__ Vt) {
  long long i = (long long)blockIdx.x * 256 + threadIdx.x;
  long long total = (long long)B * NH * HD * SP;
  if (i >= total) return;
  int s = (int)(i % SP);
  long long r = i / SP;
  int d = (int)(r % HD); r /= HD;
  int h = (int)(r % NH);
  int b = (int)(r / NH);
  Vt[i] = (s < S) ? qkv[((long long)b * S + s) * TD + 2 * D + h * HD + d] : (__bf16)0.f;
}

// ---------------- flash attention, region stack ----------------
// grid(B*NH, QT), block 32 (1 wave). One 16-query tile; keys in chunks of 32.
__global__ void k_attn_region(const __bf16* __restrict__ qkv, const __bf16* __restrict__ Vt,
                              __bf16* __restrict__ attn) {
  int b = blockIdx.x / NH, h = blockIdx.x % NH;
  int q0 = blockIdx.y * 16;
  int lane = threadIdx.x & 31, lr = lane & 15, hi = lane >> 4;

  int qrow = q0 + lr; if (qrow > S - 1) qrow = S - 1;
  const __bf16* qp = qkv + ((long long)b * S + qrow) * TD + h * HD;
  v16bf qf0 = load_frag(qp, 0, hi);
  v16bf qf1 = load_frag(qp, 32, hi);

  float mrow[8], lrow[8];
  v8f o[4] = {};
#pragma unroll
  for (int j = 0; j < 8; ++j) { mrow[j] = -3e38f; lrow[j] = 0.f; }

  __shared__ __bf16 P[16 * 32];
  const __bf16* vtb = Vt + (long long)(b * NH + h) * HD * SP;

  for (int kc = 0; kc < SP; kc += 32) {
    int k0 = kc + lr, k1 = kc + 16 + lr;
    int kr0 = k0 > S - 1 ? S - 1 : k0;
    int kr1 = k1 > S - 1 ? S - 1 : k1;
    const __bf16* kp0 = qkv + ((long long)b * S + kr0) * TD + D + h * HD;
    const __bf16* kp1 = qkv + ((long long)b * S + kr1) * TD + D + h * HD;
    v8f s0 = {}, s1 = {};
    s0 = wmma_bf16(qf0, load_frag(kp0, 0, hi), s0);
    s0 = wmma_bf16(qf1, load_frag(kp0, 32, hi), s0);
    s1 = wmma_bf16(qf0, load_frag(kp1, 0, hi), s1);
    s1 = wmma_bf16(qf1, load_frag(kp1, 32, hi), s1);

    bool v0 = k0 < S, v1 = k1 < S;
    float rm[8];
#pragma unroll
    for (int j = 0; j < 8; ++j) {
      float a0 = v0 ? s0[j] * 0.125f : -3e38f;
      float a1 = v1 ? s1[j] * 0.125f : -3e38f;
      s0[j] = a0; s1[j] = a1;
      rm[j] = fmaxf(a0, a1);
    }
#pragma unroll
    for (int off = 1; off < 16; off <<= 1)
#pragma unroll
      for (int j = 0; j < 8; ++j) rm[j] = fmaxf(rm[j], __shfl_xor(rm[j], off, 32));

    float p0e[8], p1e[8], rs[8], sc[8];
#pragma unroll
    for (int j = 0; j < 8; ++j) {
      float mn = fmaxf(mrow[j], rm[j]);
      sc[j] = __expf(mrow[j] - mn);
      p0e[j] = __expf(s0[j] - mn);
      p1e[j] = __expf(s1[j] - mn);
      rs[j] = p0e[j] + p1e[j];
      mrow[j] = mn;
    }
#pragma unroll
    for (int off = 1; off < 16; off <<= 1)
#pragma unroll
      for (int j = 0; j < 8; ++j) rs[j] += __shfl_xor(rs[j], off, 32);
#pragma unroll
    for (int j = 0; j < 8; ++j) lrow[j] = lrow[j] * sc[j] + rs[j];
#pragma unroll
    for (int t = 0; t < 4; ++t)
#pragma unroll
      for (int j = 0; j < 8; ++j) o[t][j] *= sc[j];

    // C-layout -> A-layout via LDS
#pragma unroll
    for (int j = 0; j < 8; ++j) {
      P[(j + 8 * hi) * 32 + lr]      = (__bf16)p0e[j];
      P[(j + 8 * hi) * 32 + lr + 16] = (__bf16)p1e[j];
    }
    __syncthreads();
    v16bf pf = load_frag(&P[lr * 32], 0, hi);
#pragma unroll
    for (int t = 0; t < 4; ++t) {
      const __bf16* vp = vtb + (long long)(t * 16 + lr) * SP + kc;
      v16bf vf = load_frag(vp, 0, hi);
      o[t] = wmma_bf16(pf, vf, o[t]);
    }
    __syncthreads();
  }

  float inv[8];
#pragma unroll
  for (int j = 0; j < 8; ++j) inv[j] = 1.f / lrow[j];
#pragma unroll
  for (int t = 0; t < 4; ++t)
#pragma unroll
    for (int j = 0; j < 8; ++j) {
      int s = q0 + j + 8 * hi;
      if (s < S)
        attn[((long long)b * S + s) * D + h * HD + t * 16 + lr] = (__bf16)(o[t][j] * inv[j]);
    }
}

// ---------------- attention-pool logits + top-k (softmax is monotone -> top-k of logits) ----------
__global__ void k_topk(const float* __restrict__ a, const float* __restrict__ wv,
                       int* __restrict__ idxabs) {
  int b = blockIdx.x, tid = threadIdx.x;     // 256 threads
  __shared__ float lg[S];
  for (int s = tid; s < S; s += 256) {
    const float* row = a + ((long long)b * S + s) * D;
    float dot = 0.f;
    for (int d = 0; d < D; ++d) dot += row[d] * wv[d];
    lg[s] = dot;
  }
  __syncthreads();
  __shared__ float bvs[256];
  __shared__ int   bis[256];
  for (int r = 0; r < TOPK; ++r) {
    float best = -3e38f; int bidx = 0x7fffffff;
    for (int s = tid; s < S; s += 256) {
      float v = lg[s];
      if (v > best) { best = v; bidx = s; }
    }
    bvs[tid] = best; bis[tid] = bidx;
    __syncthreads();
    for (int st = 128; st > 0; st >>= 1) {
      if (tid < st) {
        if (bvs[tid + st] > bvs[tid] ||
            (bvs[tid + st] == bvs[tid] && bis[tid + st] < bis[tid])) {
          bvs[tid] = bvs[tid + st]; bis[tid] = bis[tid + st];
        }
      }
      __syncthreads();
    }
    if (tid == 0) { idxabs[b * TOPK + r] = b * S + bis[0]; lg[bis[0]] = -3e38f; }
    __syncthreads();
  }
}

// ---------------- token-stack attention: one 16x16 tile per (b,h), 1 wave ----------------
__global__ void k_attn_token(const __bf16* __restrict__ qkvt, __bf16* __restrict__ attnt) {
  int b = blockIdx.x / NH, h = blockIdx.x % NH;
  int lane = threadIdx.x & 31, lr = lane & 15, hi = lane >> 4;
  int base = b * TOPK;

  const __bf16* qp = qkvt + (long long)(base + lr) * TD + h * HD;
  v16bf qf0 = load_frag(qp, 0, hi);
  v16bf qf1 = load_frag(qp, 32, hi);
  const __bf16* kp = qkvt + (long long)(base + lr) * TD + D + h * HD;
  v8f s0 = {};
  s0 = wmma_bf16(qf0, load_frag(kp, 0, hi), s0);
  s0 = wmma_bf16(qf1, load_frag(kp, 32, hi), s0);

  float rm[8], pe[8], rs[8];
#pragma unroll
  for (int j = 0; j < 8; ++j) { s0[j] *= 0.125f; rm[j] = s0[j]; }
#pragma unroll
  for (int off = 1; off < 16; off <<= 1)
#pragma unroll
    for (int j = 0; j < 8; ++j) rm[j] = fmaxf(rm[j], __shfl_xor(rm[j], off, 32));
#pragma unroll
  for (int j = 0; j < 8; ++j) { pe[j] = __expf(s0[j] - rm[j]); rs[j] = pe[j]; }
#pragma unroll
  for (int off = 1; off < 16; off <<= 1)
#pragma unroll
    for (int j = 0; j < 8; ++j) rs[j] += __shfl_xor(rs[j], off, 32);

  __shared__ __bf16 P[16 * 32];
  __shared__ __bf16 Vs[64 * 32];
#pragma unroll
  for (int j = 0; j < 8; ++j) {
    P[(j + 8 * hi) * 32 + lr]      = (__bf16)pe[j];
    P[(j + 8 * hi) * 32 + lr + 16] = (__bf16)0.f;   // zero-pad keys 16..31
  }
  for (int e = lane; e < 64 * 16; e += 32) {
    int d = e >> 4, key = e & 15;
    Vs[d * 32 + key]      = qkvt[(long long)(base + key) * TD + 2 * D + h * HD + d];
    Vs[d * 32 + key + 16] = (__bf16)0.f;
  }
  __syncthreads();

  v16bf pf = load_frag(&P[lr * 32], 0, hi);
#pragma unroll
  for (int t = 0; t < 4; ++t) {
    v16bf vf = load_frag(&Vs[(t * 16 + lr) * 32], 0, hi);
    v8f o = {};
    o = wmma_bf16(pf, vf, o);
#pragma unroll
    for (int j = 0; j < 8; ++j) {
      int srow = base + j + 8 * hi;
      attnt[(long long)srow * D + h * HD + t * 16 + lr] = (__bf16)(o[j] / rs[j]);
    }
  }
}

__global__ void k_out(const float* __restrict__ at, float* __restrict__ out) {
  int i = blockIdx.x * 256 + threadIdx.x;
  if (i >= B * D) return;
  int b = i / D, d = i % D;
  out[i] = at[(long long)(b * TOPK) * D + d];
}

// ---------------- host orchestration ----------------
extern "C" void kernel_launch(void* const* d_in, const int* in_sizes, int n_in,
                              void* d_out, int out_size, void* d_ws, size_t ws_size,
                              hipStream_t stream) {
  (void)in_sizes; (void)n_in; (void)out_size; (void)ws_size;
  const float* x      = (const float*)d_in[0];
  const float* cls    = (const float*)d_in[1];
  const float* ln_g   = (const float*)d_in[2];
  const float* ln_b   = (const float*)d_in[3];
  const float* wqkv_r = (const float*)d_in[4];
  const float* bqkv_r = (const float*)d_in[5];
  const float* wo_r   = (const float*)d_in[6];
  const float* bo_r   = (const float*)d_in[7];
  const float* w_attw = (const float*)d_in[8];
  // d_in[9] = b_attw: constant shift, does not change top-k ordering
  const float* wqkv_t = (const float*)d_in[10];
  const float* bqkv_t = (const float*)d_in[11];
  const float* wo_t   = (const float*)d_in[12];
  const float* bo_t   = (const float*)d_in[13];
  float* out = (float*)d_out;

  char* w = (char*)d_ws;
  size_t off = 0;
  auto alloc = [&](size_t bytes) -> void* {
    void* p = w + off;
    off = (off + bytes + 255) & ~(size_t)255;
    return p;
  };
  float*  yf    = (float*) alloc((size_t)MS * D * 4);
  float*  z0f   = (float*) alloc((size_t)MS * D * 4);
  __bf16* z0b   = (__bf16*)alloc((size_t)MS * D * 2);
  float*  af    = (float*) alloc((size_t)MS * D * 4);
  __bf16* ab    = (__bf16*)alloc((size_t)MS * D * 2);
  float*  projf = (float*) alloc((size_t)MS * D * 4);
  __bf16* qkvb  = (__bf16*)alloc((size_t)MS * TD * 2);
  __bf16* attnb = (__bf16*)alloc((size_t)MS * D * 2);
  __bf16* vtb   = (__bf16*)alloc((size_t)B * NH * HD * SP * 2);
  __bf16* wqr   = (__bf16*)alloc((size_t)L * TD * D * 2);
  __bf16* wor   = (__bf16*)alloc((size_t)L * D * D * 2);
  __bf16* wqt   = (__bf16*)alloc((size_t)L * TD * D * 2);
  __bf16* wot   = (__bf16*)alloc((size_t)L * D * D * 2);
  int*    idxA  = (int*)   alloc((size_t)MT * 4);
  float*  z0tf  = (float*) alloc((size_t)MT * D * 4);
  __bf16* z0tb  = (__bf16*)alloc((size_t)MT * D * 2);
  float*  atf   = (float*) alloc((size_t)MT * D * 4);
  __bf16* atb   = (__bf16*)alloc((size_t)MT * D * 2);
  float*  ptf   = (float*) alloc((size_t)MT * D * 4);
  __bf16* qktb  = (__bf16*)alloc((size_t)MT * TD * 2);
  __bf16* attb  = (__bf16*)alloc((size_t)MT * D * 2);

  auto cvt = [&](const float* src, __bf16* dst, long long n) {
    k_cvt_bf16<<<(unsigned)((n + 255) / 256), 256, 0, stream>>>(src, dst, n);
  };
  cvt(wqkv_r, wqr, (long long)L * TD * D);
  cvt(wo_r,   wor, (long long)L * D * D);
  cvt(wqkv_t, wqt, (long long)L * TD * D);
  cvt(wo_t,   wot, (long long)L * D * D);

  long long ytot = (long long)B * S * D;
  k_build_y<<<(unsigned)((ytot + 255) / 256), 256, 0, stream>>>(x, cls, yf);
  k_ln<<<MS, 128, 0, stream>>>(yf, nullptr, nullptr, ln_g, ln_b, z0f, z0b);

  // ---- region stack ----
  const __bf16* cur = z0b;
  long long vtot = (long long)B * NH * HD * SP;
  for (int l = 0; l < L; ++l) {
    dim3 gq((MS + 127) / 128, TD / 64);
    k_gemm<<<gq, 128, 0, stream>>>(cur, wqr + (size_t)l * TD * D, bqkv_r + l * TD,
                                   nullptr, qkvb, MS, TD);
    k_vtr<<<(unsigned)((vtot + 255) / 256), 256, 0, stream>>>(qkvb, vtb);
    k_attn_region<<<dim3(B * NH, QT), 32, 0, stream>>>(qkvb, vtb, attnb);
    dim3 go((MS + 127) / 128, D / 64);
    k_gemm<<<go, 128, 0, stream>>>(attnb, wor + (size_t)l * D * D, bo_r + l * D,
                                   projf, nullptr, MS, D);
    k_ln<<<MS, 128, 0, stream>>>(projf, z0f, nullptr, ln_g, ln_b, af, ab);
    cur = ab;
  }

  // ---- top-k + gather + LN ----
  k_topk<<<B, 256, 0, stream>>>(af, w_attw, idxA);
  k_ln<<<MT, 128, 0, stream>>>(yf, nullptr, idxA, ln_g, ln_b, z0tf, z0tb);

  // ---- token stack ----
  const __bf16* curt = z0tb;
  for (int l = 0; l < L; ++l) {
    dim3 gq(1, TD / 64);
    k_gemm<<<gq, 128, 0, stream>>>(curt, wqt + (size_t)l * TD * D, bqkv_t + l * TD,
                                   nullptr, qktb, MT, TD);
    k_attn_token<<<B * NH, 32, 0, stream>>>(qktb, attb);
    dim3 go(1, D / 64);
    k_gemm<<<go, 128, 0, stream>>>(attb, wot + (size_t)l * D * D, bo_t + l * D,
                                   ptf, nullptr, MT, D);
    k_ln<<<MT, 128, 0, stream>>>(ptf, z0tf, nullptr, ln_g, ln_b, atf, atb);
    curt = atb;
  }

  k_out<<<(B * D + 255) / 256, 256, 0, stream>>>(atf, out);
}
